// AttentionLayer_9096740733682
// MI455X (gfx1250) — compile-verified, matrix operand order checked by reading, not run on hardware
//
#include <hip/hip_runtime.h>
#include <type_traits>

typedef __attribute__((ext_vector_type(16))) _Float16 v16h;
typedef __attribute__((ext_vector_type(8)))  _Float16 v8h;
typedef __attribute__((ext_vector_type(4)))  _Float16 v4h;
typedef __attribute__((ext_vector_type(2)))  _Float16 v2h;
typedef __attribute__((ext_vector_type(8)))  float    v8f;
typedef __attribute__((ext_vector_type(4)))  float    v4f;

#define MODEL_DIM 512
#define NUM_HEADS 8
#define HEAD_DIM  64
#define BATCH     8
#define SEQ       2048
#define NROWS     (BATCH * SEQ)        // 16384
#define NEG_BIG   (-1.0e30f)
#define LOG2E     1.44269504088896340736f

// ---------------------------------------------------------------------------
// Universal WMMA fragment loader for V_WMMA_F32_16X16X32_F16 (wave32).
// Per ISA 7.12.2 a lane's 16 f16 = two contiguous 8-elem runs at
//   row + (lane>>4)*8  and  row + 16 + (lane>>4)*8  ->  2x b128 loads.
// ---------------------------------------------------------------------------
__device__ __forceinline__ v16h load_frag_rm(const _Float16* base, int ld, int lane) {
  const _Float16* row = base + (lane & 15) * ld + (lane >> 4) * 8;
  v8h lo = *(const v8h*)(row);
  v8h hi = *(const v8h*)(row + 16);
  v16h f;
#pragma unroll
  for (int i = 0; i < 8; ++i) { f[i] = lo[i]; f[8 + i] = hi[i]; }
  return f;
}

__device__ __forceinline__ v8f wmma_f16(v16h a, v16h b, v8f c) {
  return __builtin_amdgcn_wmma_f32_16x16x32_f16(false, a, false, b, (short)0, c,
                                                false, false);
}

// Raw v_max_num_f32 (no NaN-quieting canonicalize pairs).
__device__ __forceinline__ float fmax_raw(float a, float b) {
  float d;
  asm("v_max_num_f32 %0, %1, %2" : "=v"(d) : "v"(a), "v"(b));
  return d;
}

// ---------------------------------------------------------------------------
// 16-lane-group reductions: DPP row_ror fused directly into the ALU op.
// ---------------------------------------------------------------------------
#define DPP_STEP_MAX(x, N)                                                     \
  {                                                                            \
    float t_;                                                                  \
    asm("v_max_num_f32_e64_dpp %0, %1, %1 row_ror:" #N                         \
        " row_mask:0xf bank_mask:0xf bound_ctrl:1"                             \
        : "=v"(t_) : "v"(x));                                                  \
    x = t_;                                                                    \
  }
#define DPP_STEP_ADD(x, N)                                                     \
  {                                                                            \
    float t_;                                                                  \
    asm("v_add_f32_e64_dpp %0, %1, %1 row_ror:" #N                             \
        " row_mask:0xf bank_mask:0xf bound_ctrl:1"                             \
        : "=v"(t_) : "v"(x));                                                  \
    x = t_;                                                                    \
  }
__device__ __forceinline__ float rmax16(float x) {
  DPP_STEP_MAX(x, 1) DPP_STEP_MAX(x, 2) DPP_STEP_MAX(x, 4) DPP_STEP_MAX(x, 8)
  return x;
}
__device__ __forceinline__ float rsum16(float x) {
  DPP_STEP_ADD(x, 1) DPP_STEP_ADD(x, 2) DPP_STEP_ADD(x, 4) DPP_STEP_ADD(x, 8)
  return x;
}

// ---------------------------------------------------------------------------
// Kernel 1: projection, templated on Z (0=Q scaled, 1=K, 2=V transposed).
// ---------------------------------------------------------------------------
#define LDS_STRIDE 40   // 40 f16 = 80 B rows, 16B aligned
#define CS_STRIDE  72   // 72 f16 = 144 B rows, 16B aligned

template <int Z>
__global__ __launch_bounds__(256) void proj_kernel(
    const float* __restrict__ X, const float* __restrict__ W,
    const float* __restrict__ bias, _Float16* __restrict__ Out) {
  __shared__ alignas(16) _Float16 As[64 * LDS_STRIDE];    // [row][k]
  __shared__ alignas(16) _Float16 BsT[64 * LDS_STRIDE];   // [n][k] (transposed)

  const int rowBase = blockIdx.x * 64;
  const int colBase = blockIdx.y * 64;
  const int tid = threadIdx.x;
  const int wid = tid >> 5, lane = tid & 31;
  const int wave_m = wid & 3, wave_n = wid >> 2;

  v8f acc0 = {}, acc1 = {};
#pragma unroll 1
  for (int k0 = 0; k0 < MODEL_DIM; k0 += 32) {
    __syncthreads();
#pragma unroll
    for (int j = 0; j < 2; ++j) {           // A tile 64x32 f32->f16
      int e = (tid + j * 256) * 4;
      int r = e >> 5, c = e & 31;
      v4f x = *(const v4f*)(X + (size_t)(rowBase + r) * MODEL_DIM + k0 + c);
      v4h h4;
      h4[0] = (_Float16)x[0]; h4[1] = (_Float16)x[1];
      h4[2] = (_Float16)x[2]; h4[3] = (_Float16)x[3];
      *(v4h*)(&As[r * LDS_STRIDE + c]) = h4;
    }
    {                                       // B tile 32x64 -> BsT[n][k], paired
      int r2 = (tid >> 4) * 2;
      int c4 = (tid & 15) * 4;
      const float* w0 = W + (size_t)(k0 + r2) * MODEL_DIM + colBase + c4;
      v4f wa = *(const v4f*)w0;
      v4f wb = *(const v4f*)(w0 + MODEL_DIM);
#pragma unroll
      for (int q = 0; q < 4; ++q) {
        v2h p; p[0] = (_Float16)wa[q]; p[1] = (_Float16)wb[q];
        *(v2h*)(&BsT[(c4 + q) * LDS_STRIDE + r2]) = p;
      }
    }
    __syncthreads();
    v16h a  = load_frag_rm(&As[(wave_m * 16) * LDS_STRIDE], LDS_STRIDE, lane);
    v16h b0 = load_frag_rm(&BsT[(wave_n * 32) * LDS_STRIDE], LDS_STRIDE, lane);
    v16h b1 = load_frag_rm(&BsT[(wave_n * 32 + 16) * LDS_STRIDE], LDS_STRIDE, lane);
    acc0 = wmma_f16(a, b0, acc0);
    acc1 = wmma_f16(a, b1, acc1);
  }

  const int N = lane & 15, half = lane >> 4;
  const int h = colBase >> 6;               // block spans exactly one head
  // Q carries 1/sqrt(64) AND log2(e) so attention can use raw v_exp (exp2).
  constexpr float sc = (Z == 0) ? 0.125f * LOG2E : 1.0f;

  if constexpr (Z == 2) {
    // V transposed [b][h][d][l]: lane's 8 acc rows = 8 consecutive tokens
    const int l0 = rowBase + wave_m * 16 + half * 8;
    const int bb = l0 >> 11, l = l0 & (SEQ - 1);
#pragma unroll
    for (int t = 0; t < 2; ++t) {
      int d = wave_n * 32 + t * 16 + N;
      float bs = bias[colBase + d];
      v8h o;
#pragma unroll
      for (int r = 0; r < 8; ++r)
        o[r] = (_Float16)(((t == 0) ? acc0[r] : acc1[r]) + bs);
      size_t idx = ((((size_t)bb * NUM_HEADS + h) * HEAD_DIM) + d) * SEQ + l;
      *(v8h*)(Out + idx) = o;
    }
  } else {
    // Q/K [b][h][l][d]: restage 64x64 C tile in LDS -> coalesced b128 stores
    __shared__ alignas(16) _Float16 Cs[64 * CS_STRIDE];
    __syncthreads();
#pragma unroll
    for (int r = 0; r < 8; ++r) {
      int M = r + 8 * half;
#pragma unroll
      for (int t = 0; t < 2; ++t) {
        int c = wave_n * 32 + t * 16 + N;
        float val = (((t == 0) ? acc0[r] : acc1[r]) + bias[colBase + c]) * sc;
        Cs[(wave_m * 16 + M) * CS_STRIDE + c] = (_Float16)val;
      }
    }
    __syncthreads();
#pragma unroll
    for (int j = 0; j < 2; ++j) {
      int cid = tid + j * 256;
      int row = cid >> 3, c = (cid & 7) * 8;
      int gr = rowBase + row;
      int bb = gr >> 11, l = gr & (SEQ - 1);
      v8h v = *(const v8h*)(&Cs[row * CS_STRIDE + c]);
      *(v8h*)(Out + ((((size_t)bb * NUM_HEADS + h) * SEQ) + l) * HEAD_DIM + c) = v;
    }
  }
}

// ---------------------------------------------------------------------------
// Kernel 2: fused causal flash attention, 64-key chunks (base-2 softmax).
//   grid = (SEQ/64, B*H), block = 128 (4 independent waves x 16 query rows).
//   16 wmma per softmax pass; per-pass bookkeeping amortized over 64 keys.
// ---------------------------------------------------------------------------
__global__ __launch_bounds__(128) void flash_attn_kernel(
    const _Float16* __restrict__ Q, const _Float16* __restrict__ K,
    const _Float16* __restrict__ Vt, _Float16* __restrict__ O) {
  __shared__ alignas(16) _Float16 Ws[4][16 * CS_STRIDE];  // P staging / O restage

  const int bh = blockIdx.y;
  const int bb = bh >> 3, h = bh & 7;
  const int tid = threadIdx.x;
  const int wid = tid >> 5, lane = tid & 31;
  const int qrow = blockIdx.x * 64 + wid * 16;   // wave's first query row

  const _Float16* Qbh = Q + (size_t)bh * SEQ * HEAD_DIM;   // [l][d]
  const _Float16* Kbh = K + (size_t)bh * SEQ * HEAD_DIM;   // [l][d]
  const _Float16* Vbh = Vt + (size_t)bh * HEAD_DIM * SEQ;  // [d][l]

  v16h aQ0 = load_frag_rm(Qbh + (size_t)qrow * HEAD_DIM + 0, HEAD_DIM, lane);
  v16h aQ1 = load_frag_rm(Qbh + (size_t)qrow * HEAD_DIM + 32, HEAD_DIM, lane);

  v8f o0 = {}, o1 = {}, o2 = {}, o3 = {};
  float m[8], lsum[8];
#pragma unroll
  for (int r = 0; r < 8; ++r) { m[r] = NEG_BIG; lsum[r] = 0.0f; }

  const int N = lane & 15, half = lane >> 4;

  auto chunk = [&](int kc, auto maskc) {     // processes keys [kc, kc+64)
    constexpr bool MASK = decltype(maskc)::value;
    __builtin_prefetch(Kbh + (size_t)(kc + 64 + lane) * HEAD_DIM, 0, 1);
    __builtin_prefetch(Vbh + (size_t)lane * SEQ + kc + 64, 0, 1);

    // S = Q @ K^T : four 16-key tiles, K-dim 64 chained
    v8f s[4];
#pragma unroll
    for (int t = 0; t < 4; ++t) {
      v8f st = {};
      st = wmma_f16(aQ0, load_frag_rm(Kbh + (size_t)(kc + 16 * t) * HEAD_DIM + 0,
                                      HEAD_DIM, lane), st);
      st = wmma_f16(aQ1, load_frag_rm(Kbh + (size_t)(kc + 16 * t) * HEAD_DIM + 32,
                                      HEAD_DIM, lane), st);
      s[t] = st;
    }

    // base-2 online softmax; per-pass cost amortized over 64 keys
#pragma unroll
    for (int r = 0; r < 8; ++r) {
      int M = r + 8 * half;
      if constexpr (MASK) {
        int qg = qrow + M;
#pragma unroll
        for (int t = 0; t < 4; ++t)
          if (kc + 16 * t + N > qg) s[t][r] = NEG_BIG;
      }
      float mx = fmax_raw(fmax_raw(s[0][r], s[1][r]), fmax_raw(s[2][r], s[3][r]));
      mx = rmax16(mx);
      float mnew  = fmax_raw(m[r], mx);
      float alpha = __builtin_amdgcn_exp2f(m[r] - mnew);
      float p0 = __builtin_amdgcn_exp2f(s[0][r] - mnew);
      float p1 = __builtin_amdgcn_exp2f(s[1][r] - mnew);
      float p2 = __builtin_amdgcn_exp2f(s[2][r] - mnew);
      float p3 = __builtin_amdgcn_exp2f(s[3][r] - mnew);
      float rs = rsum16((p0 + p1) + (p2 + p3));
      lsum[r] = lsum[r] * alpha + rs;
      m[r] = mnew;
      o0[r] *= alpha; o1[r] *= alpha; o2[r] *= alpha; o3[r] *= alpha;
      _Float16* row = &Ws[wid][M * CS_STRIDE];
      row[N]      = (_Float16)p0;
      row[N + 16] = (_Float16)p1;
      row[N + 32] = (_Float16)p2;
      row[N + 48] = (_Float16)p3;
    }
    asm volatile("s_wait_dscnt 0" ::: "memory");   // wave-private LDS RAW
    v16h aP0 = load_frag_rm(&Ws[wid][0],  CS_STRIDE, lane);   // keys kc..+31
    v16h aP1 = load_frag_rm(&Ws[wid][32], CS_STRIDE, lane);   // keys kc+32..+63

    // O += P @ V  (B-frag row n = d, K-run = keys, contiguous in V^T)
    o0 = wmma_f16(aP0, load_frag_rm(Vbh + (size_t)0 * SEQ + kc, SEQ, lane), o0);
    o1 = wmma_f16(aP0, load_frag_rm(Vbh + (size_t)16 * SEQ + kc, SEQ, lane), o1);
    o2 = wmma_f16(aP0, load_frag_rm(Vbh + (size_t)32 * SEQ + kc, SEQ, lane), o2);
    o3 = wmma_f16(aP0, load_frag_rm(Vbh + (size_t)48 * SEQ + kc, SEQ, lane), o3);
    o0 = wmma_f16(aP1, load_frag_rm(Vbh + (size_t)0 * SEQ + kc + 32, SEQ, lane), o0);
    o1 = wmma_f16(aP1, load_frag_rm(Vbh + (size_t)16 * SEQ + kc + 32, SEQ, lane), o1);
    o2 = wmma_f16(aP1, load_frag_rm(Vbh + (size_t)32 * SEQ + kc + 32, SEQ, lane), o2);
    o3 = wmma_f16(aP1, load_frag_rm(Vbh + (size_t)48 * SEQ + kc + 32, SEQ, lane), o3);
  };

  const int fullChunks = qrow >> 6;                 // kc+63 <= qrow : no mask
  const int nchunks = (qrow + 79) >> 6;             // keys [0, qrow+16)
#pragma unroll 1
  for (int ch = 0; ch < fullChunks; ++ch)
    chunk(ch * 64, std::integral_constant<bool, false>{});
#pragma unroll 1
  for (int ch = fullChunks; ch < nchunks; ++ch)     // exactly 1 diagonal chunk
    chunk(ch * 64, std::integral_constant<bool, true>{});

  // epilogue: normalize, restage wave tile in LDS, coalesced b128 stores
#pragma unroll
  for (int r = 0; r < 8; ++r) {
    int M = r + 8 * half;
    float inv = 1.0f / lsum[r];
    _Float16* row = &Ws[wid][M * CS_STRIDE];
    row[N]      = (_Float16)(o0[r] * inv);
    row[N + 16] = (_Float16)(o1[r] * inv);
    row[N + 32] = (_Float16)(o2[r] * inv);
    row[N + 48] = (_Float16)(o3[r] * inv);
  }
  asm volatile("s_wait_dscnt 0" ::: "memory");      // wave-private LDS RAW
  {
    int row = lane >> 1, c0 = (lane & 1) * 32;      // 2 lanes per row
    int token = qrow + row;
    size_t gbase = ((size_t)bb * SEQ + token) * MODEL_DIM + h * HEAD_DIM + c0;
#pragma unroll
    for (int j = 0; j < 4; ++j) {
      v8h v = *(const v8h*)(&Ws[wid][row * CS_STRIDE + c0 + j * 8]);
      *(v8h*)(O + gbase + j * 8) = v;
    }
  }
}

// ---------------------------------------------------------------------------
// Kernel 3: output projection. out = O(f16) @ Wo + bo -> f32.
// ---------------------------------------------------------------------------
#define CSF_STRIDE 68   // 68 f32 = 272 B rows, 16B aligned
__global__ __launch_bounds__(256) void out_proj_kernel(
    const _Float16* __restrict__ Oin, const float* __restrict__ Wo,
    const float* __restrict__ bo, float* __restrict__ out) {
  __shared__ alignas(16) _Float16 As[64 * LDS_STRIDE];
  __shared__ alignas(16) _Float16 BsT[64 * LDS_STRIDE];
  __shared__ alignas(16) float    CsF[64 * CSF_STRIDE];

  const int rowBase = blockIdx.x * 64;
  const int colBase = blockIdx.y * 64;
  const int tid = threadIdx.x;
  const int wid = tid >> 5, lane = tid & 31;
  const int wave_m = wid & 3, wave_n = wid >> 2;

  v8f acc0 = {}, acc1 = {};
#pragma unroll 1
  for (int k0 = 0; k0 < MODEL_DIM; k0 += 32) {
    __syncthreads();
    {
      int e = tid * 8;
      int r = e >> 5, c = e & 31;
      v8h a8 = *(const v8h*)(Oin + (size_t)(rowBase + r) * MODEL_DIM + k0 + c);
      *(v8h*)(&As[r * LDS_STRIDE + c]) = a8;
    }
    {
      int r2 = (tid >> 4) * 2;
      int c4 = (tid & 15) * 4;
      const float* w0 = Wo + (size_t)(k0 + r2) * MODEL_DIM + colBase + c4;
      v4f wa = *(const v4f*)w0;
      v4f wb = *(const v4f*)(w0 + MODEL_DIM);
#pragma unroll
      for (int q = 0; q < 4; ++q) {
        v2h p; p[0] = (_Float16)wa[q]; p[1] = (_Float16)wb[q];
        *(v2h*)(&BsT[(c4 + q) * LDS_STRIDE + r2]) = p;
      }
    }
    __syncthreads();
    v16h a  = load_frag_rm(&As[(wave_m * 16) * LDS_STRIDE], LDS_STRIDE, lane);
    v16h b0 = load_frag_rm(&BsT[(wave_n * 32) * LDS_STRIDE], LDS_STRIDE, lane);
    v16h b1 = load_frag_rm(&BsT[(wave_n * 32 + 16) * LDS_STRIDE], LDS_STRIDE, lane);
    acc0 = wmma_f16(a, b0, acc0);
    acc1 = wmma_f16(a, b1, acc1);
  }

  const int N = lane & 15, half = lane >> 4;
  __syncthreads();
#pragma unroll
  for (int r = 0; r < 8; ++r) {
    int M = r + 8 * half;
#pragma unroll
    for (int t = 0; t < 2; ++t) {
      int c = wave_n * 32 + t * 16 + N;
      CsF[(wave_m * 16 + M) * CSF_STRIDE + c] =
          ((t == 0) ? acc0[r] : acc1[r]) + bo[colBase + c];
    }
  }
  __syncthreads();
#pragma unroll
  for (int j = 0; j < 4; ++j) {
    int cid = tid + j * 256;
    int row = cid >> 4, c = (cid & 15) * 4;
    v4f v = *(const v4f*)(&CsF[row * CSF_STRIDE + c]);
    *(v4f*)(out + (size_t)(rowBase + row) * MODEL_DIM + colBase + c) = v;
  }
}

// ---------------------------------------------------------------------------
extern "C" void kernel_launch(void* const* d_in, const int* in_sizes, int n_in,
                              void* d_out, int out_size, void* d_ws, size_t ws_size,
                              hipStream_t stream) {
  const float* query = (const float*)d_in[0];
  const float* key   = (const float*)d_in[1];
  const float* value = (const float*)d_in[2];
  const float* Wq = (const float*)d_in[3];
  const float* bq = (const float*)d_in[4];
  const float* Wk = (const float*)d_in[5];
  const float* bk = (const float*)d_in[6];
  const float* Wv = (const float*)d_in[7];
  const float* bv = (const float*)d_in[8];
  const float* Wo = (const float*)d_in[9];
  const float* bo = (const float*)d_in[10];
  float* out = (float*)d_out;

  const size_t elems = (size_t)NROWS * MODEL_DIM;   // 16 MB f16 each
  _Float16* Qh = (_Float16*)d_ws;
  _Float16* Kh = Qh + elems;
  _Float16* Vh = Kh + elems;    // transposed [b][h][d][l]
  _Float16* Oh = Vh + elems;

  dim3 gGemm(NROWS / 64, MODEL_DIM / 64);
  proj_kernel<0><<<gGemm, 256, 0, stream>>>(query, Wq, bq, Qh);
  proj_kernel<1><<<gGemm, 256, 0, stream>>>(key,   Wk, bk, Kh);
  proj_kernel<2><<<gGemm, 256, 0, stream>>>(value, Wv, bv, Vh);

  dim3 gAttn(SEQ / 64, BATCH * NUM_HEADS);
  flash_attn_kernel<<<gAttn, 128, 0, stream>>>(Qh, Kh, Vh, Oh);

  dim3 gOut(NROWS / 64, MODEL_DIM / 64);
  out_proj_kernel<<<gOut, 256, 0, stream>>>(Oh, Wo, bo, out);
}
